// maskGRU_51376398794893
// MI455X (gfx1250) — compile-verified, matrix operand order checked by reading, not run on hardware
//
#include <hip/hip_runtime.h>
#include <hip/hip_bf16.h>
#include <cmath>

// ---------------- problem constants ----------------
#define TT   512
#define BB   64
#define HH   1024
#define EE   256
#define OTD  300          // real OT
#define OTP  320          // OT padded to multiple of 32 for WMMA K / N tiling
#define OCD  256
#define KW   1792         // fused K: 2E (512) + E (256, scattered W3) + H (1024)
#define NN   4096         // fused N: 3H (pre) + H (hh3 duplicate)
#define TCH  64           // timestep chunk for output projections
#define OUTS_STRIDE (TT*BB*OCD)   // 8388608 floats per output head

typedef __attribute__((ext_vector_type(16))) __bf16 v16bf;
typedef __attribute__((ext_vector_type(8)))  __bf16 v8bf;
typedef __attribute__((ext_vector_type(8)))  float  v8f;

union FragBF { v16bf v; v8bf h[2]; };

#define WMMA_BF16(A, B, C) __builtin_amdgcn_wmma_f32_16x16x32_bf16( \
    false, (A), false, (B), (short)0, (C), false, false)

// ---------------------------------------------------------------------------
// Prep kernels
// ---------------------------------------------------------------------------

__device__ __forceinline__ int w3row(int n) {
    // x12 columns [512:1024) += x3[:,0:512); [1536:2048) += x3[:,512:1024);
    // [2560:3072) += x3[:,1024:1536)
    if (n >= 512  && n < 1024) return n - 512;
    if (n >= 1536 && n < 2048) return n - 1024;
    if (n >= 2560 && n < 3072) return n - 1536;
    return -1;
}

__global__ void prep_wcat(const float* __restrict__ W12,  // [3072,512]
                          const float* __restrict__ W3,   // [1536,256]
                          const float* __restrict__ Wh,   // [3072,1024]
                          __bf16* __restrict__ Wcat)      // [NN,KW]
{
    int k = blockIdx.x * blockDim.x + threadIdx.x;
    int n = blockIdx.y;
    if (k >= KW) return;
    float v = 0.0f;
    if (k < 512) {
        if (n < 3072) v = W12[n * 512 + k];
    } else if (k < 768) {
        if (n < 3072) {
            int r = w3row(n);
            if (r >= 0) v = W3[r * 256 + (k - 512)];
        }
    } else {
        int hr = (n < 3072) ? n : (n - 1024);   // rows 3072..4095 -> Wh rows 2048..3071
        v = Wh[hr * 1024 + (k - 768)];
    }
    Wcat[(size_t)n * KW + k] = (__bf16)v;
}

__global__ void prep_bcat(const float* __restrict__ b12,
                          const float* __restrict__ b3,
                          const float* __restrict__ bh,
                          float* __restrict__ bcat)
{
    int n = blockIdx.x * blockDim.x + threadIdx.x;
    if (n >= NN) return;
    float v;
    if (n < 3072) {
        v = b12[n] + bh[n];
        int r = w3row(n);
        if (r >= 0) v += b3[r];
    } else {
        v = bh[n - 1024];
    }
    bcat[n] = v;
}

__global__ void prep_emb(const float* __restrict__ E1,
                         const float* __restrict__ E2,
                         const float* __restrict__ E3,
                         __bf16* __restrict__ Ebf)   // 3 tables back-to-back
{
    int idx = blockIdx.x * blockDim.x + threadIdx.x;   // 0..3*65536-1
    int which = idx >> 16;
    int off   = idx & 0xFFFF;
    const float* src = (which == 0) ? E1 : (which == 1) ? E2 : E3;
    Ebf[idx] = (__bf16)src[off];
}

__global__ void prep_wo(const float* __restrict__ Wo1,  // [300,512]
                        const float* __restrict__ Wo2,
                        __bf16* __restrict__ WoP)       // 2 x [OTP,512]
{
    int k = blockIdx.x * blockDim.x + threadIdx.x;      // 0..511
    int y = blockIdx.y;                                 // sel*OTP + n
    int sel = y / OTP, n = y % OTP;
    const float* W = sel ? Wo2 : Wo1;
    float v = (n < OTD) ? W[n * 512 + k] : 0.0f;
    WoP[((size_t)sel * OTP + n) * 512 + k] = (__bf16)v;
}

__global__ void prep_wout(const float* __restrict__ Wout1, // [256,300]
                          const float* __restrict__ Wout2,
                          __bf16* __restrict__ WoutP)      // 2 x [256,OTP]
{
    int k = blockIdx.x * blockDim.x + threadIdx.x;         // 0..OTP-1
    int y = blockIdx.y;                                    // sel*256 + n
    if (k >= OTP) return;
    int sel = y >> 8, n = y & 255;
    const float* W = sel ? Wout2 : Wout1;
    float v = (k < OTD) ? W[n * OTD + k] : 0.0f;
    WoutP[((size_t)sel * 256 + n) * OTP + k] = (__bf16)v;
}

__global__ void prep_h(const float* __restrict__ h0,
                       float* __restrict__ hf, __bf16* __restrict__ hbf)
{
    int idx = blockIdx.x * blockDim.x + threadIdx.x;       // 0..65535
    float v = h0[idx];
    hf[idx] = v;
    hbf[idx] = (__bf16)v;
}

// ---------------------------------------------------------------------------
// Per-step fused GEMM: PRE[64,NN] = [emb1|emb2|emb3|h] (64xKW) x Wcat^T + bcat
// 256 waves: 4 M-tiles x 64 N-groups (each group = 4 consecutive 16-col tiles,
// reusing one A fragment for 4 WMMAs). K loop split into 4 straight-line
// segments (E1 | E2 | E3 | h) -> no per-iteration segment branches.
// ---------------------------------------------------------------------------
__global__ __launch_bounds__(256) void step_gemm(
    const __bf16* __restrict__ Wcat, const float* __restrict__ bcat,
    const __bf16* __restrict__ E1bf, const __bf16* __restrict__ E2bf,
    const __bf16* __restrict__ E3bf,
    const int* __restrict__ id1, const int* __restrict__ id2,
    const int* __restrict__ id3,
    const __bf16* __restrict__ hbf,   // [64,1024]
    float* __restrict__ PRE, int t)
{
    const int lane  = threadIdx.x & 31;
    const int wave  = blockIdx.x * 8 + (threadIdx.x >> 5);
    const int mtile = wave & 3;
    const int ngrp  = wave >> 2;
    const int m0 = mtile * 16;
    const int n0 = ngrp * 64;
    const int lrow = lane & 15;
    const int koff = (lane >> 4) * 8;      // K sub-offset inside 32-wide tile
    const int mrow = m0 + lrow;

    // per-segment A row base pointers (pre-offset by koff)
    const __bf16* paE1 = E1bf + (size_t)id1[t * BB + mrow] * 256 + koff;
    const __bf16* paE2 = E2bf + (size_t)id2[t * BB + mrow] * 256 + koff;
    const __bf16* paE3 = E3bf + (size_t)id3[t * BB + mrow] * 256 + koff;
    const __bf16* paH  = hbf + (size_t)mrow * 1024 + koff;

    // four B row pointers (pre-offset by koff), advanced 32 elems per K step
    const __bf16* pb0 = Wcat + (size_t)(n0 +  0 + lrow) * KW + koff;
    const __bf16* pb1 = Wcat + (size_t)(n0 + 16 + lrow) * KW + koff;
    const __bf16* pb2 = Wcat + (size_t)(n0 + 32 + lrow) * KW + koff;
    const __bf16* pb3 = Wcat + (size_t)(n0 + 48 + lrow) * KW + koff;

    v8f acc0 = {}, acc1 = {}, acc2 = {}, acc3 = {};

#define K_STEP(PA)                                                          \
    {                                                                       \
        FragBF a, b0, b1, b2, b3;                                           \
        a.h[0]  = *(const v8bf*)(PA);                                       \
        a.h[1]  = *(const v8bf*)(PA + 16);                                  \
        b0.h[0] = *(const v8bf*)(pb0);                                      \
        b0.h[1] = *(const v8bf*)(pb0 + 16);                                 \
        b1.h[0] = *(const v8bf*)(pb1);                                      \
        b1.h[1] = *(const v8bf*)(pb1 + 16);                                 \
        b2.h[0] = *(const v8bf*)(pb2);                                      \
        b2.h[1] = *(const v8bf*)(pb2 + 16);                                 \
        b3.h[0] = *(const v8bf*)(pb3);                                      \
        b3.h[1] = *(const v8bf*)(pb3 + 16);                                 \
        acc0 = WMMA_BF16(a.v, b0.v, acc0);                                  \
        acc1 = WMMA_BF16(a.v, b1.v, acc1);                                  \
        acc2 = WMMA_BF16(a.v, b2.v, acc2);                                  \
        acc3 = WMMA_BF16(a.v, b3.v, acc3);                                  \
        PA += 32; pb0 += 32; pb1 += 32; pb2 += 32; pb3 += 32;               \
    }

#pragma unroll 2
    for (int kl = 0; kl < 256; kl += 32) K_STEP(paE1)
#pragma unroll 2
    for (int kl = 0; kl < 256; kl += 32) K_STEP(paE2)
#pragma unroll 2
    for (int kl = 0; kl < 256; kl += 32) K_STEP(paE3)
#pragma unroll 2
    for (int kl = 0; kl < 1024; kl += 32) K_STEP(paH)
#undef K_STEP

    const int hi = (lane >> 4) * 8;
    v8f accs[4] = {acc0, acc1, acc2, acc3};
#pragma unroll
    for (int j = 0; j < 4; ++j) {
        int ncol = n0 + j * 16 + lrow;
        float bv = bcat[ncol];
#pragma unroll
        for (int r = 0; r < 8; ++r)
            PRE[(size_t)(m0 + hi + r) * NN + ncol] = accs[j][r] + bv;
    }
}

// ---------------------------------------------------------------------------
// Per-step elementwise GRU update.
// pre cols: [0:H) g1-pre, [H:2H) g2-pre, [2H:3H) pre3=x3+hh3, [3H:4H) hh3
// ---------------------------------------------------------------------------
__global__ void step_elem(const float* __restrict__ PRE,
                          float* __restrict__ hf, __bf16* __restrict__ hbf,
                          __bf16* __restrict__ hs_chunk, int tl)
{
    int idx = blockIdx.x * blockDim.x + threadIdx.x;   // 0..65535
    int m = idx >> 10;
    int j = idx & 1023;
    const float* row = PRE + (size_t)m * NN;
    float g1   = 1.0f / (1.0f + __expf(-row[j]));
    float g2   = 1.0f / (1.0f + __expf(-row[1024 + j]));
    float pre3 = row[2048 + j];
    float hh3  = row[3072 + j];
    float x3   = pre3 - hh3;
    float tr   = tanhf(g2 * hh3 + x3);
    float hn   = g1 * hf[idx] + (1.0f - g1) * tr;
    hf[idx]  = hn;
    hbf[idx] = (__bf16)hn;
    hs_chunk[(size_t)tl * (BB * HH) + idx] = (__bf16)hn;
}

// ---------------------------------------------------------------------------
// Output GEMM 1: O = relu(Hs_chunk[:, sel*512 : sel*512+512] x WoP^T + bo)
// M = TCH*BB = 4096, K = 512, N = OTP(320). blockIdx.y = sel (head).
// ---------------------------------------------------------------------------
__global__ __launch_bounds__(256) void out_gemm1(
    const __bf16* __restrict__ hs_chunk,   // [4096,1024]
    const __bf16* __restrict__ WoP,        // 2 x [OTP,512]
    const float* __restrict__ bo1, const float* __restrict__ bo2,
    __bf16* __restrict__ Obf)              // 2 x [4096,OTP]
{
    const int sel  = blockIdx.y;
    const int lane = threadIdx.x & 31;
    const int wave = blockIdx.x * 8 + (threadIdx.x >> 5);
    const int mtile = wave / (OTP / 16);   // 256 m-tiles
    const int ntile = wave % (OTP / 16);   // 20 n-tiles
    const int m0 = mtile * 16, n0 = ntile * 16;
    const int lrow = lane & 15;
    const int koff = (lane >> 4) * 8;
    const float* bo  = sel ? bo2 : bo1;

    const __bf16* pa = hs_chunk + (size_t)(m0 + lrow) * 1024 + sel * 512 + koff;
    const __bf16* pb = WoP + (size_t)sel * OTP * 512
                     + (size_t)(n0 + lrow) * 512 + koff;

    v8f acc = {};
#pragma unroll 2
    for (int kb = 0; kb < 512; kb += 32) {
        FragBF a, b;
        a.h[0] = *(const v8bf*)(pa);
        a.h[1] = *(const v8bf*)(pa + 16);
        b.h[0] = *(const v8bf*)(pb);
        b.h[1] = *(const v8bf*)(pb + 16);
        acc = WMMA_BF16(a.v, b.v, acc);
        pa += 32; pb += 32;
    }
    const int hi = (lane >> 4) * 8;
    const int ncol = n0 + lrow;
    float bv = (ncol < OTD) ? bo[ncol] : 0.0f;
    __bf16* out = Obf + (size_t)sel * 4096 * OTP;
#pragma unroll
    for (int r = 0; r < 8; ++r) {
        float v = (ncol < OTD) ? fmaxf(acc[r] + bv, 0.0f) : 0.0f;
        out[(size_t)(m0 + hi + r) * OTP + ncol] = (__bf16)v;
    }
}

// ---------------------------------------------------------------------------
// Output GEMM 2: OUTS = O x WoutP^T + bout -> d_out slice for this chunk.
// M = 4096, K = OTP(320), N = 256. blockIdx.y = sel.
// ---------------------------------------------------------------------------
__global__ __launch_bounds__(256) void out_gemm2(
    const __bf16* __restrict__ Obf,        // 2 x [4096,OTP]
    const __bf16* __restrict__ WoutP,      // 2 x [256,OTP]
    const float* __restrict__ bout1, const float* __restrict__ bout2,
    float* __restrict__ dout, int chunk)
{
    const int sel  = blockIdx.y;
    const int lane = threadIdx.x & 31;
    const int wave = blockIdx.x * 8 + (threadIdx.x >> 5);
    const int mtile = wave >> 4;           // 256 m-tiles
    const int ntile = wave & 15;           // 16 n-tiles
    const int m0 = mtile * 16, n0 = ntile * 16;
    const int lrow = lane & 15;
    const int koff = (lane >> 4) * 8;
    const float* bout = sel ? bout2 : bout1;

    const __bf16* pa = Obf + (size_t)sel * 4096 * OTP
                     + (size_t)(m0 + lrow) * OTP + koff;
    const __bf16* pb = WoutP + (size_t)sel * 256 * OTP
                     + (size_t)(n0 + lrow) * OTP + koff;

    v8f acc = {};
#pragma unroll
    for (int kb = 0; kb < OTP; kb += 32) {
        FragBF a, b;
        a.h[0] = *(const v8bf*)(pa);
        a.h[1] = *(const v8bf*)(pa + 16);
        b.h[0] = *(const v8bf*)(pb);
        b.h[1] = *(const v8bf*)(pb + 16);
        acc = WMMA_BF16(a.v, b.v, acc);
        pa += 32; pb += 32;
    }
    const int hi = (lane >> 4) * 8;
    const int ncol = n0 + lrow;
    float bv = bout[ncol];
    float* base = dout + (size_t)sel * OUTS_STRIDE + (size_t)chunk * 4096 * OCD;
#pragma unroll
    for (int r = 0; r < 8; ++r)
        base[(size_t)(m0 + hi + r) * OCD + ncol] = acc[r] + bv;
}

__global__ void copy_hT(const float* __restrict__ hf, float* __restrict__ dout)
{
    int idx = blockIdx.x * blockDim.x + threadIdx.x;   // 0..65535
    dout[(size_t)2 * OUTS_STRIDE + idx] = hf[idx];
}

// ---------------------------------------------------------------------------
extern "C" void kernel_launch(void* const* d_in, const int* in_sizes, int n_in,
                              void* d_out, int out_size, void* d_ws, size_t ws_size,
                              hipStream_t stream)
{
    const int*   id1   = (const int*)  d_in[0];
    const int*   id2   = (const int*)  d_in[1];
    const int*   id3   = (const int*)  d_in[2];
    const float* E1    = (const float*)d_in[3];
    const float* E2    = (const float*)d_in[4];
    const float* E3    = (const float*)d_in[5];
    const float* W12   = (const float*)d_in[6];
    const float* b12   = (const float*)d_in[7];
    const float* W3    = (const float*)d_in[8];
    const float* b3    = (const float*)d_in[9];
    const float* Wh    = (const float*)d_in[10];
    const float* bh    = (const float*)d_in[11];
    const float* Wo1   = (const float*)d_in[12];
    const float* bo1   = (const float*)d_in[13];
    const float* Wo2   = (const float*)d_in[14];
    const float* bo2   = (const float*)d_in[15];
    const float* Wout1 = (const float*)d_in[16];
    const float* bout1 = (const float*)d_in[17];
    const float* Wout2 = (const float*)d_in[18];
    const float* bout2 = (const float*)d_in[19];
    const float* h0    = (const float*)d_in[20];
    float* dout = (float*)d_out;

    // ---- carve workspace (~31 MB) ----
    char* p = (char*)d_ws;
    auto carve = [&](size_t bytes) {
        char* r = p; p += (bytes + 255) & ~(size_t)255; return r;
    };
    __bf16* Wcat   = (__bf16*)carve((size_t)NN * KW * 2);          // 14.7 MB
    __bf16* Ebf    = (__bf16*)carve((size_t)3 * 65536 * 2);        // 384 KB
    __bf16* WoP    = (__bf16*)carve((size_t)2 * OTP * 512 * 2);    // 640 KB
    __bf16* WoutP  = (__bf16*)carve((size_t)2 * 256 * OTP * 2);    // 320 KB
    float*  bcat   = (float*) carve((size_t)NN * 4);
    float*  hf     = (float*) carve((size_t)BB * HH * 4);
    __bf16* hbf    = (__bf16*)carve((size_t)BB * HH * 2);
    float*  PRE    = (float*) carve((size_t)BB * NN * 4);          // 1 MB
    __bf16* hsch   = (__bf16*)carve((size_t)TCH * BB * HH * 2);    // 8 MB
    __bf16* Obf    = (__bf16*)carve((size_t)2 * 4096 * OTP * 2);   // 5 MB

    // ---- prep ----
    prep_wcat <<<dim3(7, NN),      256, 0, stream>>>(W12, W3, Wh, Wcat);
    prep_bcat <<<NN / 256,         256, 0, stream>>>(b12, b3, bh, bcat);
    prep_emb  <<<3 * 65536 / 256,  256, 0, stream>>>(E1, E2, E3, Ebf);
    prep_wo   <<<dim3(2, 2 * OTP), 256, 0, stream>>>(Wo1, Wo2, WoP);
    prep_wout <<<dim3(2, 2 * 256), 256, 0, stream>>>(Wout1, Wout2, WoutP);
    prep_h    <<<BB * HH / 256,    256, 0, stream>>>(h0, hf, hbf);

    const __bf16* E1bf = Ebf;
    const __bf16* E2bf = Ebf + 65536;
    const __bf16* E3bf = Ebf + 2 * 65536;

    // ---- recurrence, chunked for output projections ----
    for (int c = 0; c < TT / TCH; ++c) {
        for (int tl = 0; tl < TCH; ++tl) {
            int t = c * TCH + tl;
            step_gemm<<<32, 256, 0, stream>>>(Wcat, bcat, E1bf, E2bf, E3bf,
                                              id1, id2, id3, hbf, PRE, t);
            step_elem<<<BB * HH / 256, 256, 0, stream>>>(PRE, hf, hbf, hsch, tl);
        }
        out_gemm1<<<dim3(256 * (OTP / 16) / 8, 2), 256, 0, stream>>>(
            hsch, WoP, bo1, bo2, Obf);
        out_gemm2<<<dim3(256 * 16 / 8, 2), 256, 0, stream>>>(
            Obf, WoutP, bout1, bout2, dout, c);
    }
    copy_hT<<<BB * HH / 256, 256, 0, stream>>>(hf, dout);
}
